// OverlapTransformerLayer_28217935134735
// MI455X (gfx1250) — matrix operand order repaired and checked by reading, not compiled, not run on hardware
//
#include <hip/hip_runtime.h>
#include <hip/hip_bf16.h>
#include <math.h>

// ---------------------------------------------------------------------------
// Problem constants (from reference)
// ---------------------------------------------------------------------------
constexpr int Bc   = 2;
constexpr int Sc   = 2048;
constexpr int Hc   = 2048;
constexpr int NHc  = 16;
constexpr int KVHc = 4;
constexpr int HDc  = 128;
constexpr int Gc   = NHc / KVHc;              // 4
constexpr int Pc   = KVHc * (Gc + 2) * HDc;   // 3072
constexpr int Ec   = 8;
constexpr int Fc   = 1024;
constexpr int Tc   = Bc * Sc;                 // 4096 tokens
constexpr int MAXT = 4096;                    // max tokens per expert
constexpr float QSCALE = 0.08838834764831845f; // 1/sqrt(128)

typedef __attribute__((ext_vector_type(16))) __bf16 v16bf;
typedef __attribute__((ext_vector_type(8)))  __bf16 v8bf;
typedef __attribute__((ext_vector_type(8)))  float  v8f;

static __device__ inline v16bf cat8(v8bf lo, v8bf hi) {
  return __builtin_shufflevector(lo, hi, 0,1,2,3,4,5,6,7,8,9,10,11,12,13,14,15);
}
static __device__ inline v8f wmma_bf16(v16bf a, v16bf b, v8f c) {
  // D = A(16x32 bf16) x B(32x16 bf16) + C(16x16 f32)
  return __builtin_amdgcn_wmma_f32_16x16x32_bf16(false, a, false, b, (short)0, c,
                                                 false, false);
}
static __device__ inline v8f vzero8() {
  v8f v = {0.f,0.f,0.f,0.f,0.f,0.f,0.f,0.f};
  return v;
}
// Reductions across a 16-lane half-group (matches WMMA C layout: row lives
// across lanes 0..15 / 16..31 of one VGPR).
static __device__ inline float halfmax(float v) {
  #pragma unroll
  for (int o = 1; o < 16; o <<= 1) v = fmaxf(v, __shfl_xor(v, o, 32));
  return v;
}
static __device__ inline float halfsum(float v) {
  #pragma unroll
  for (int o = 1; o < 16; o <<= 1) v += __shfl_xor(v, o, 32);
  return v;
}

// ---------------------------------------------------------------------------
// CDNA5 async global->LDS copy (tracked with ASYNCcnt).
// VDST holds the LDS-relative byte address (per ISA 08_async_tensor).
// ---------------------------------------------------------------------------
static __device__ inline uint32_t lds_off_u32(const void* p) {
  return (uint32_t)(uintptr_t)(__attribute__((address_space(3))) const void*)p;
}
static __device__ inline void async_b128(uint32_t lds, const void* g) {
  asm volatile("global_load_async_to_lds_b128 %0, %1, off"
               :: "v"(lds), "v"(g)
               : "memory");
}
static __device__ inline void wait_async0() {
  asm volatile("s_wait_asynccnt 0x0" ::: "memory");
}

// ---------------------------------------------------------------------------
// Weight conversion kernels (f32 -> bf16, with transposes so every GEMM has
// a contiguous-K "B" operand: C[m,n] = sum_k A[m,k]*B[n,k])
// ---------------------------------------------------------------------------
__global__ __launch_bounds__(256) void cvt_kernel(const float* __restrict__ s,
                                                  __bf16* __restrict__ d, size_t n) {
  size_t i = (size_t)blockIdx.x * 256 + threadIdx.x;
  if (i < n) d[i] = (__bf16)s[i];
}
// w1 (H, E*F) -> w1t[e][f][h]
__global__ __launch_bounds__(256) void cvt_w1_kernel(const float* __restrict__ w1,
                                                     __bf16* __restrict__ w1t) {
  size_t i = (size_t)blockIdx.x * 256 + threadIdx.x;
  if (i >= (size_t)Ec * Fc * Hc) return;
  int h = (int)(i % Hc);
  size_t ef = i / Hc;
  int f = (int)(ef % Fc);
  int e = (int)(ef / Fc);
  w1t[i] = (__bf16)w1[(size_t)h * (Ec * Fc) + (size_t)e * Fc + f];
}
// w2 (E*F, H) -> w2t[e][h][f]
__global__ __launch_bounds__(256) void cvt_w2_kernel(const float* __restrict__ w2,
                                                     __bf16* __restrict__ w2t) {
  size_t i = (size_t)blockIdx.x * 256 + threadIdx.x;
  if (i >= (size_t)Ec * Hc * Fc) return;
  int f = (int)(i % Fc);
  size_t eh = i / Fc;
  int h = (int)(eh % Hc);
  int e = (int)(eh / Hc);
  w2t[i] = (__bf16)w2[((size_t)e * Fc + f) * Hc + h];
}

// ---------------------------------------------------------------------------
// LayerNorm 1: x (f32) -> xn (bf16)
// ---------------------------------------------------------------------------
__global__ __launch_bounds__(256) void ln1_kernel(const float* __restrict__ x,
                                                  const float* __restrict__ w,
                                                  const float* __restrict__ b,
                                                  __bf16* __restrict__ xn) {
  __shared__ float rsum[256], rsq[256];
  const int row = blockIdx.x, tid = threadIdx.x;
  const float* xr = x + (size_t)row * Hc;
  float v[8], s = 0.f, ss = 0.f;
  #pragma unroll
  for (int i = 0; i < 8; ++i) {
    v[i] = xr[tid + i * 256];
    s += v[i]; ss += v[i] * v[i];
  }
  rsum[tid] = s; rsq[tid] = ss;
  __syncthreads();
  for (int o = 128; o > 0; o >>= 1) {
    if (tid < o) { rsum[tid] += rsum[tid + o]; rsq[tid] += rsq[tid + o]; }
    __syncthreads();
  }
  const float mean = rsum[0] * (1.0f / Hc);
  const float var  = rsq[0] * (1.0f / Hc) - mean * mean;
  const float rstd = rsqrtf(var + 1e-5f);
  #pragma unroll
  for (int i = 0; i < 8; ++i) {
    int c = tid + i * 256;
    xn[(size_t)row * Hc + c] = (__bf16)((v[i] - mean) * rstd * w[c] + b[c]);
  }
}

// ---------------------------------------------------------------------------
// LayerNorm 2 + router softmax + top-2, also copies x1 into d_out.
// ---------------------------------------------------------------------------
__global__ __launch_bounds__(256) void ln2_router_kernel(
    const float* __restrict__ x1, const float* __restrict__ w,
    const float* __restrict__ b, const float* __restrict__ rw,
    __bf16* __restrict__ xn2, float* __restrict__ outc,
    int* __restrict__ tki, float* __restrict__ tkp) {
  __shared__ float rsum[256], rsq[256];
  __shared__ float lr[256 * 8];
  __shared__ float lg[8];
  const int row = blockIdx.x, tid = threadIdx.x;
  const float* xr = x1 + (size_t)row * Hc;
  float v[8], s = 0.f, ss = 0.f;
  #pragma unroll
  for (int i = 0; i < 8; ++i) {
    int c = tid + i * 256;
    v[i] = xr[c];
    outc[(size_t)row * Hc + c] = v[i];   // d_out = x1 (MoE adds on top)
    s += v[i]; ss += v[i] * v[i];
  }
  rsum[tid] = s; rsq[tid] = ss;
  __syncthreads();
  for (int o = 128; o > 0; o >>= 1) {
    if (tid < o) { rsum[tid] += rsum[tid + o]; rsq[tid] += rsq[tid + o]; }
    __syncthreads();
  }
  const float mean = rsum[0] * (1.0f / Hc);
  const float var  = rsq[0] * (1.0f / Hc) - mean * mean;
  const float rstd = rsqrtf(var + 1e-5f);
  float pe[8] = {0,0,0,0,0,0,0,0};
  #pragma unroll
  for (int i = 0; i < 8; ++i) {
    int c = tid + i * 256;
    float nv = (v[i] - mean) * rstd * w[c] + b[c];
    xn2[(size_t)row * Hc + c] = (__bf16)nv;
    #pragma unroll
    for (int e = 0; e < 8; ++e) pe[e] += nv * rw[(size_t)c * Ec + e];
  }
  #pragma unroll
  for (int e = 0; e < 8; ++e) lr[tid * 8 + e] = pe[e];
  __syncthreads();
  if (tid < 8) {
    float a = 0.f;
    for (int j = 0; j < 256; ++j) a += lr[j * 8 + tid];
    lg[tid] = a;
  }
  __syncthreads();
  if (tid == 0) {
    float mx = lg[0];
    #pragma unroll
    for (int e = 1; e < 8; ++e) mx = fmaxf(mx, lg[e]);
    float p[8], se = 0.f;
    #pragma unroll
    for (int e = 0; e < 8; ++e) { p[e] = __expf(lg[e] - mx); se += p[e]; }
    #pragma unroll
    for (int e = 0; e < 8; ++e) p[e] /= se;
    int i0 = 0;
    #pragma unroll
    for (int e = 1; e < 8; ++e) if (p[e] > p[i0]) i0 = e;
    int i1 = (i0 == 0) ? 1 : 0;
    #pragma unroll
    for (int e = 0; e < 8; ++e) if (e != i0 && p[e] > p[i1]) i1 = e;
    tki[row * 2 + 0] = i0; tki[row * 2 + 1] = i1;
    tkp[row * 2 + 0] = p[i0]; tkp[row * 2 + 1] = p[i1];
  }
}

// ---------------------------------------------------------------------------
// Split QKV + reorder: qkv f32 (T x 3072) -> q bf16 [B,NH,S,HD] (pre-scaled),
// k bf16 [B,KVH,S,HD], vT bf16 [B,KVH,HD,S]
// ---------------------------------------------------------------------------
__global__ __launch_bounds__(256) void reorder_kernel(const float* __restrict__ qkv,
                                                      __bf16* __restrict__ qb,
                                                      __bf16* __restrict__ kb,
                                                      __bf16* __restrict__ vtb) {
  size_t i = (size_t)blockIdx.x * 256 + threadIdx.x;
  if (i >= (size_t)Tc * Pc) return;
  int p = (int)(i % Pc);
  size_t t = i / Pc;
  int bb = (int)(t / Sc), s2 = (int)(t % Sc);
  int kvh = p / ((Gc + 2) * HDc);
  int rem = p % ((Gc + 2) * HDc);
  int g = rem / HDc, d = rem % HDc;
  float v = qkv[i];
  if (g < Gc) {
    int hh = kvh * Gc + g;
    qb[(((size_t)bb * NHc + hh) * Sc + s2) * HDc + d] = (__bf16)(v * QSCALE);
  } else if (g == Gc) {
    kb[(((size_t)bb * KVHc + kvh) * Sc + s2) * HDc + d] = (__bf16)v;
  } else {
    vtb[(((size_t)bb * KVHc + kvh) * HDc + d) * Sc + s2] = (__bf16)v;
  }
}

// ---------------------------------------------------------------------------
// Tiled WMMA GEMM: C[m,n] = sum_k A[m,k] * B[n,k]   (both bf16, K contiguous)
// Block = 256 threads (8 waves). Tile 128x128, K-step 32.
// LDS tiles are double-buffered and filled with GLOBAL_LOAD_ASYNC_TO_LDS_B128,
// so stage(i+1) overlaps the 8 WMMAs of step i (one barrier per K-step).
// MODE 0: C f32                      (QKV projection)
// MODE 1: C f32 = acc + residual     (attention out-proj + residual)
// MODE 2: MoE up: A rows gathered via token list, gelu, bf16 out
// MODE 3: MoE down: scatter atomicAdd(out[token], gate*acc)
// ---------------------------------------------------------------------------
template <int MODE>
__global__ __launch_bounds__(256) void gemm_kernel(
    const __bf16* __restrict__ A, const __bf16* __restrict__ Bm,
    float* __restrict__ Cf, const float* __restrict__ Xres,
    __bf16* __restrict__ Hout, float* __restrict__ Oacc,
    const int* __restrict__ cnts, const int* __restrict__ list,
    const float* __restrict__ gate, int M, int N, int K) {
  constexpr int BK = 32, LS = 48;  // LDS row stride (elems): 96B, 32B-aligned
  __shared__ __align__(32) __bf16 As[2][128 * LS];
  __shared__ __align__(32) __bf16 Bs[2][128 * LS];

  const int tid  = threadIdx.x;
  const int lane = tid & 31, wave = tid >> 5;
  const int half = lane >> 4, l16 = lane & 15;
  const int wm = wave & 3, wn = wave >> 2;

  int e = 0, cnt = M;
  const int m0 = blockIdx.x * 128;
  const int n0 = blockIdx.y * 128;
  if (MODE >= 2) {
    e = blockIdx.z;
    cnt = cnts[e];
    if (m0 >= cnt) return;
  }

  const __bf16* Aptr = A;
  const __bf16* Bptr = Bm;
  if (MODE == 2) Bptr = Bm + (size_t)e * N * K;
  if (MODE == 3) { Aptr = A + (size_t)e * MAXT * K; Bptr = Bm + (size_t)e * N * K; }

  v8f acc[2][4];
  #pragma unroll
  for (int mi = 0; mi < 2; ++mi)
    #pragma unroll
    for (int ni = 0; ni < 4; ++ni) acc[mi][ni] = vzero8();

  const int grow = tid >> 1;         // 0..127 (tile row this thread stages)
  const int gseg = (tid & 1) * 16;   // k segment 0 / 16

  size_t arow;
  if (MODE == 2) {
    int r = m0 + grow;
    if (r >= cnt) r = cnt - 1;       // clamp; rows past cnt are discarded
    arow = (size_t)list[e * MAXT + r];
  } else {
    arow = (size_t)(m0 + grow);
  }
  const size_t abase = arow * (size_t)K + gseg;
  const size_t bbase = (size_t)(n0 + grow) * (size_t)K + gseg;

  // Per-thread LDS destination byte addresses for both buffers.
  const uint32_t laddrA[2] = { lds_off_u32(&As[0][grow * LS + gseg]),
                               lds_off_u32(&As[1][grow * LS + gseg]) };
  const uint32_t laddrB[2] = { lds_off_u32(&Bs[0][grow * LS + gseg]),
                               lds_off_u32(&Bs[1][grow * LS + gseg]) };

  auto stage = [&](int buf, int k0) {
    async_b128(laddrA[buf],      Aptr + abase + k0);
    async_b128(laddrA[buf] + 16, Aptr + abase + k0 + 8);
    async_b128(laddrB[buf],      Bptr + bbase + k0);
    async_b128(laddrB[buf] + 16, Bptr + bbase + k0 + 8);
  };

  stage(0, 0);
  const int nk = K / BK;
  for (int i = 0; i < nk; ++i) {
    const int buf = i & 1;
    wait_async0();        // this wave's LDS tile for step i has landed
    __syncthreads();      // all waves' tiles landed; prev buffer free
    if (i + 1 < nk) stage(buf ^ 1, (i + 1) * BK);   // overlaps WMMAs below

    const __bf16* Asb = &As[buf][0];
    const __bf16* Bsb = &Bs[buf][0];
    v16bf afrag[2], bfrag[4];
    #pragma unroll
    for (int mi = 0; mi < 2; ++mi) {
      int r = wm * 32 + mi * 16 + l16;
      v8bf lo = *(const v8bf*)&Asb[r * LS + half * 16];
      v8bf hi = *(const v8bf*)&Asb[r * LS + half * 16 + 8];
      afrag[mi] = cat8(lo, hi);
    }
    #pragma unroll
    for (int ni = 0; ni < 4; ++ni) {
      int r = wn * 64 + ni * 16 + l16;
      bfrag[ni] = *(const v16bf*)&Bsb[r * LS + half * 16];
    }
    #pragma unroll
    for (int mi = 0; mi < 2; ++mi)
      #pragma unroll
      for (int ni = 0; ni < 4; ++ni)
        acc[mi][ni] = wmma_bf16(afrag[mi], bfrag[ni], acc[mi][ni]);
  }

  // Epilogue (WMMA C layout: n = lane&15, m = r + 8*(lane>>4))
  #pragma unroll
  for (int mi = 0; mi < 2; ++mi) {
    const int lmb = wm * 32 + mi * 16 + half * 8;
    #pragma unroll
    for (int ni = 0; ni < 4; ++ni) {
      const int n = n0 + wn * 64 + ni * 16 + l16;
      v8f a = acc[mi][ni];
      #pragma unroll
      for (int r = 0; r < 8; ++r) {
        const int lrow = lmb + r;
        const float v = a[r];
        if (MODE == 0) {
          Cf[(size_t)(m0 + lrow) * N + n] = v;
        } else if (MODE == 1) {
          size_t idx = (size_t)(m0 + lrow) * N + n;
          Cf[idx] = v + Xres[idx];
        } else if (MODE == 2) {
          int mrow = m0 + lrow;
          if (mrow < cnt) {
            float g = 0.5f * v * (1.0f + erff(v * 0.7071067811865476f));
            Hout[((size_t)e * MAXT + mrow) * N + n] = (__bf16)g;
          }
        } else {
          int mrow = m0 + lrow;
          if (mrow < cnt) {
            int tok = list[e * MAXT + mrow];
            float g = gate[e * MAXT + mrow];
            atomicAdd(&Oacc[(size_t)tok * N + n], g * v);
          }
        }
      }
    }
  }
}

// ---------------------------------------------------------------------------
// Flash attention: one wave per (b, head, 16-query tile). Q pre-scaled.
// scores via WMMA QK^T (K contiguous in d), PV via WMMA with V transposed.
// ---------------------------------------------------------------------------
__global__ __launch_bounds__(32) void attn_kernel(const __bf16* __restrict__ qb,
                                                  const __bf16* __restrict__ kb,
                                                  const __bf16* __restrict__ vtb,
                                                  __bf16* __restrict__ aob) {
  __shared__ __align__(32) __bf16 pm[16][32];
  const int lane = threadIdx.x;
  const int half = lane >> 4, l16 = lane & 15;
  const int s0 = blockIdx.x * 16;
  const int bh = blockIdx.y;            // b*NH + head
  const int bb = bh >> 4, head = bh & 15;
  const int bkv = bb * KVHc + (head >> 2);

  // Load Q tile 16x128 as four 16x32 A fragments
  v16bf afrag[4];
  #pragma unroll
  for (int c = 0; c < 4; ++c) {
    const __bf16* base = qb + ((size_t)bh * Sc + s0 + l16) * HDc + c * 32;
    v8bf lo = *(const v8bf*)(base + half * 16);
    v8bf hi = *(const v8bf*)(base + half * 16 + 8);
    afrag[c] = cat8(lo, hi);
  }

  float rm[8], rs[8];
  #pragma unroll
  for (int r = 0; r < 8; ++r) { rm[r] = -1e30f; rs[r] = 0.f; }
  v8f acc_o[8];
  #pragma unroll
  for (int d = 0; d < 8; ++d) acc_o[d] = vzero8();

  for (int j = 0; j < Sc; j += 32) {
    // scores for 32 keys (two 16x16 tiles)
    v8f sc0 = vzero8(), sc1 = vzero8();
    #pragma unroll
    for (int c = 0; c < 4; ++c) {
      const __bf16* kp0 =
          kb + ((size_t)bkv * Sc + j + l16) * HDc + c * 32 + half * 16;
      sc0 = wmma_bf16(afrag[c], *(const v16bf*)kp0, sc0);
      sc1 = wmma_bf16(afrag[c], *(const v16bf*)(kp0 + 16 * HDc), sc1);
    }
    // online softmax update
    float alpha[8];
    #pragma unroll
    for (int r = 0; r < 8; ++r) {
      float tmax = halfmax(fmaxf(sc0[r], sc1[r]));
      float nm = fmaxf(rm[r], tmax);
      alpha[r] = __expf(rm[r] - nm);
      rm[r] = nm;
      float p0 = __expf(sc0[r] - nm);
      float p1 = __expf(sc1[r] - nm);
      sc0[r] = p0; sc1[r] = p1;
      rs[r] = rs[r] * alpha[r] + halfsum(p0 + p1);
    }
    // C-layout -> A-layout transpose of P through LDS (bf16)
    __syncthreads();
    #pragma unroll
    for (int r = 0; r < 8; ++r) {
      pm[half * 8 + r][l16]      = (__bf16)sc0[r];
      pm[half * 8 + r][16 + l16] = (__bf16)sc1[r];
    }
    __syncthreads();
    v8bf plo = *(const v8bf*)&pm[l16][half * 16];
    v8bf phi = *(const v8bf*)&pm[l16][half * 16 + 8];
    v16bf pfrag = cat8(plo, phi);
    // accumulate P @ V (V transposed: contiguous keys per lane)
    #pragma unroll
    for (int dt = 0; dt < 8; ++dt) {
      const __bf16* vp =
          vtb + ((size_t)bkv * HDc + dt * 16 + l16) * Sc + j + half * 16;
      v8f o = acc_o[dt];
      #pragma unroll
      for (int r = 0; r < 8; ++r) o[r] *= alpha[r];
      acc_o[dt] = wmma_bf16(pfrag, *(const v16bf*)vp, o);
    }
  }

  // write ao [B, S, NH*HD] as bf16
  #pragma unroll
  for (int dt = 0; dt < 8; ++dt)
    #pragma unroll
    for (int r = 0; r < 8; ++r) {
      int m = half * 8 + r;
      float v = acc_o[dt][r] / rs[r];
      aob[((size_t)bb * Sc + s0 + m) * (NHc * HDc) + head * HDc + dt * 16 + l16] =
          (__bf16)v;
    }
}

// ---------------------------------------------------------------------------
// MoE routing: per-expert token lists
// ---------------------------------------------------------------------------
__global__ void zero_cnt_kernel(int* cnt) {
  if (threadIdx.x < Ec) cnt[threadIdx.x] = 0;
}
__global__ __launch_bounds__(256) void build_kernel(const int* __restrict__ tki,
                                                    const float* __restrict__ tkp,
                                                    int* __restrict__ cnt,
                                                    int* __restrict__ list,
                                                    float* __restrict__ gate) {
  int t = blockIdx.x * 256 + threadIdx.x;
  if (t >= Tc) return;
  #pragma unroll
  for (int j = 0; j < 2; ++j) {
    int e = tki[t * 2 + j];
    int pos = atomicAdd(&cnt[e], 1);
    list[e * MAXT + pos] = t;
    gate[e * MAXT + pos] = tkp[t * 2 + j];
  }
}

// ---------------------------------------------------------------------------
// Host launcher
// ---------------------------------------------------------------------------
extern "C" void kernel_launch(void* const* d_in, const int* in_sizes, int n_in,
                              void* d_out, int out_size, void* d_ws, size_t ws_size,
                              hipStream_t stream) {
  const float* x      = (const float*)d_in[0];
  const float* ln1w   = (const float*)d_in[1];
  const float* ln1b   = (const float*)d_in[2];
  const float* ln2w   = (const float*)d_in[3];
  const float* ln2b   = (const float*)d_in[4];
  const float* w_qkv  = (const float*)d_in[5];
  const float* w_proj = (const float*)d_in[6];
  const float* rw     = (const float*)d_in[7];
  const float* w1     = (const float*)d_in[8];
  const float* w2     = (const float*)d_in[9];
  float* out = (float*)d_out;

  char* ws = (char*)d_ws;
  size_t off = 0;
  auto carve = [&](size_t bytes) -> char* {
    char* p = ws + off;
    off += bytes;
    off = (off + 255) & ~(size_t)255;
    return p;
  };

  __bf16* xn_bf    = (__bf16*)carve((size_t)Tc * Hc * 2);
  __bf16* wqkv_bf  = (__bf16*)carve((size_t)Pc * Hc * 2);
  __bf16* wproj_bf = (__bf16*)carve((size_t)Hc * Hc * 2);
  __bf16* w1t_bf   = (__bf16*)carve((size_t)Ec * Fc * Hc * 2);
  __bf16* w2t_bf   = (__bf16*)carve((size_t)Ec * Hc * Fc * 2);
  float*  qkv_f    = (float*)carve((size_t)Tc * Pc * 4);
  __bf16* q_bf     = (__bf16*)carve((size_t)Bc * NHc * Sc * HDc * 2);
  __bf16* k_bf     = (__bf16*)carve((size_t)Bc * KVHc * Sc * HDc * 2);
  __bf16* vt_bf    = (__bf16*)carve((size_t)Bc * KVHc * HDc * Sc * 2);
  __bf16* ao_bf    = (__bf16*)carve((size_t)Tc * Hc * 2);
  float*  x1_f     = (float*)carve((size_t)Tc * Hc * 4);
  __bf16* xn2_bf   = (__bf16*)carve((size_t)Tc * Hc * 2);
  int*    tki      = (int*)carve((size_t)Tc * 2 * 4);
  float*  tkp      = (float*)carve((size_t)Tc * 2 * 4);
  int*    cnt      = (int*)carve(Ec * 4);
  int*    list     = (int*)carve((size_t)Ec * MAXT * 4);
  float*  gate     = (float*)carve((size_t)Ec * MAXT * 4);
  __bf16* h_bf     = (__bf16*)carve((size_t)Ec * MAXT * Fc * 2);
  (void)ws_size; (void)in_sizes; (void)n_in; (void)out_size;

  // 1) weight conversions
  {
    size_t n = (size_t)Pc * Hc;
    cvt_kernel<<<(unsigned)((n + 255) / 256), 256, 0, stream>>>(w_qkv, wqkv_bf, n);
  }
  {
    size_t n = (size_t)Hc * Hc;
    cvt_kernel<<<(unsigned)((n + 255) / 256), 256, 0, stream>>>(w_proj, wproj_bf, n);
  }
  {
    size_t n = (size_t)Ec * Fc * Hc;
    cvt_w1_kernel<<<(unsigned)((n + 255) / 256), 256, 0, stream>>>(w1, w1t_bf);
    cvt_w2_kernel<<<(unsigned)((n + 255) / 256), 256, 0, stream>>>(w2, w2t_bf);
  }
  // 2) LN1
  ln1_kernel<<<Tc, 256, 0, stream>>>(x, ln1w, ln1b, xn_bf);
  // 3) QKV GEMM (4096x3072x2048)
  gemm_kernel<0><<<dim3(Tc / 128, Pc / 128), 256, 0, stream>>>(
      xn_bf, wqkv_bf, qkv_f, nullptr, nullptr, nullptr, nullptr, nullptr,
      nullptr, Tc, Pc, Hc);
  // 4) split + reorder q/k/vT
  {
    size_t n = (size_t)Tc * Pc;
    reorder_kernel<<<(unsigned)((n + 255) / 256), 256, 0, stream>>>(qkv_f, q_bf,
                                                                    k_bf, vt_bf);
  }
  // 5) flash attention
  attn_kernel<<<dim3(Sc / 16, Bc * NHc), 32, 0, stream>>>(q_bf, k_bf, vt_bf, ao_bf);
  // 6) out-projection + residual -> x1
  gemm_kernel<1><<<dim3(Tc / 128, Hc / 128), 256, 0, stream>>>(
      ao_bf, wproj_bf, x1_f, x, nullptr, nullptr, nullptr, nullptr, nullptr,
      Tc, Hc, Hc);
  // 7) LN2 + router top-2 (also seeds d_out with x1)
  ln2_router_kernel<<<Tc, 256, 0, stream>>>(x1_f, ln2w, ln2b, rw, xn2_bf, out,
                                            tki, tkp);
  // 8) expert token lists
  zero_cnt_kernel<<<1, 32, 0, stream>>>(cnt);
  build_kernel<<<(Tc + 255) / 256, 256, 0, stream>>>(tki, tkp, cnt, list, gate);
  // 9) MoE up-proj + gelu (per expert, gathered rows)
  gemm_kernel<2><<<dim3(MAXT / 128, Fc / 128, Ec), 256, 0, stream>>>(
      xn2_bf, w1t_bf, nullptr, nullptr, h_bf, nullptr, cnt, list, gate,
      MAXT, Fc, Hc);
  // 10) MoE down-proj, gated scatter-add into d_out
  gemm_kernel<3><<<dim3(MAXT / 128, Hc / 128, Ec), 256, 0, stream>>>(
      h_bf, w2t_bf, nullptr, nullptr, nullptr, out, cnt, list, gate,
      MAXT, Hc, Fc);
}
